// GlycoPeptideTreeLSTM_56049323213764
// MI455X (gfx1250) — matrix operand order repaired and checked by reading, not compiled, not run on hardware
//
#include <hip/hip_runtime.h>

// ---------------------------------------------------------------------------
// GlycoPeptide TreeLSTM — CDNA5 (gfx1250) implementation.
// All matmuls run on v_wmma_f32_16x16x32_bf16 (bf16 in, f32 accumulate).
// ---------------------------------------------------------------------------

#define BQ 512      // batch
#define LQ 48       // peptide length
#define GQ 16       // glycan nodes
#define HQ 512      // hidden
#define N4H 2048    // 4*H gate width

typedef __bf16  bf16x16 __attribute__((ext_vector_type(16)));
typedef __bf16  bf16x8  __attribute__((ext_vector_type(8)));
typedef float   f32x8   __attribute__((ext_vector_type(8)));

// ---------------- WMMA fragment loaders (layouts per CDNA5 ISA) --
// A fragment: 16x32 tile, row-major source, row m = lane&15,
//   lanes 0-15 hold K = 0..7 & 16..23 ; lanes 16-31 hold K = 8..15 & 24..31.
__device__ __forceinline__ bf16x16 frag_a(const __bf16* __restrict__ p, long ld, int lane) {
  const __bf16* row = p + (size_t)(lane & 15) * ld + ((lane >> 4) << 3);
  bf16x8 lo = *(const bf16x8*)(row);
  bf16x8 hi = *(const bf16x8*)(row + 16);
  bf16x16 f;
#pragma unroll
  for (int i = 0; i < 8; ++i) { f[i] = lo[i]; f[i + 8] = hi[i]; }
  return f;
}

// B fragment: 32x16 tile; source is W stored [N][K] row-major (we compute X@W^T),
//   column n = lane&15 ; lanes 0-15 hold K = 0..7 & 16..23 ; lanes 16-31 hold K = 8..15 & 24..31.
__device__ __forceinline__ bf16x16 frag_b(const __bf16* __restrict__ p, long ld, int lane) {
  const __bf16* row = p + (size_t)(lane & 15) * ld + ((lane >> 4) << 3);
  bf16x8 lo = *(const bf16x8*)(row);
  bf16x8 hi = *(const bf16x8*)(row + 16);
  bf16x16 f;
#pragma unroll
  for (int i = 0; i < 8; ++i) { f[i] = lo[i]; f[i + 8] = hi[i]; }
  return f;
}

// ---------------- LSTM step GEMM: Z[512][2048] = X@Wx^T + H@Wh^T ------------
// Dual K-segment (input projection + recurrent projection) fused in one pass.
// Block = 256 thr = 8 waves (2 along M, 4 along N); wave tile = 16x64.
// Grid = (512/32, 2048/256) = (16, 8).
__global__ __launch_bounds__(256)
void lstm_step_gemm_k(const __bf16* __restrict__ X, long ldx,
                      const __bf16* __restrict__ Wx, int kx,
                      const __bf16* __restrict__ H,
                      const __bf16* __restrict__ Wh,
                      float* __restrict__ Z) {
  const int lane = threadIdx.x & 31;
  const int wave = threadIdx.x >> 5;
  const int m0 = blockIdx.x * 32 + (wave & 1) * 16;
  const int n0 = blockIdx.y * 256 + (wave >> 1) * 64;

  f32x8 acc0 = {}, acc1 = {}, acc2 = {}, acc3 = {};

  // segment 1: input projection (kx is 32, 544 or 1024 — padded mult. of 32)
  for (int k0 = 0; k0 < kx; k0 += 32) {
    bf16x16 a  = frag_a(X + (size_t)m0 * ldx + k0, ldx, lane);
    bf16x16 b0 = frag_b(Wx + (size_t)(n0 +  0) * kx + k0, kx, lane);
    bf16x16 b1 = frag_b(Wx + (size_t)(n0 + 16) * kx + k0, kx, lane);
    bf16x16 b2 = frag_b(Wx + (size_t)(n0 + 32) * kx + k0, kx, lane);
    bf16x16 b3 = frag_b(Wx + (size_t)(n0 + 48) * kx + k0, kx, lane);
    acc0 = __builtin_amdgcn_wmma_f32_16x16x32_bf16(false, a, false, b0, (short)0, acc0, false, false);
    acc1 = __builtin_amdgcn_wmma_f32_16x16x32_bf16(false, a, false, b1, (short)0, acc1, false, false);
    acc2 = __builtin_amdgcn_wmma_f32_16x16x32_bf16(false, a, false, b2, (short)0, acc2, false, false);
    acc3 = __builtin_amdgcn_wmma_f32_16x16x32_bf16(false, a, false, b3, (short)0, acc3, false, false);
  }
  // segment 2: recurrent projection, K = 512
  for (int k0 = 0; k0 < HQ; k0 += 32) {
    bf16x16 a  = frag_a(H + (size_t)m0 * HQ + k0, HQ, lane);
    bf16x16 b0 = frag_b(Wh + (size_t)(n0 +  0) * HQ + k0, HQ, lane);
    bf16x16 b1 = frag_b(Wh + (size_t)(n0 + 16) * HQ + k0, HQ, lane);
    bf16x16 b2 = frag_b(Wh + (size_t)(n0 + 32) * HQ + k0, HQ, lane);
    bf16x16 b3 = frag_b(Wh + (size_t)(n0 + 48) * HQ + k0, HQ, lane);
    acc0 = __builtin_amdgcn_wmma_f32_16x16x32_bf16(false, a, false, b0, (short)0, acc0, false, false);
    acc1 = __builtin_amdgcn_wmma_f32_16x16x32_bf16(false, a, false, b1, (short)0, acc1, false, false);
    acc2 = __builtin_amdgcn_wmma_f32_16x16x32_bf16(false, a, false, b2, (short)0, acc2, false, false);
    acc3 = __builtin_amdgcn_wmma_f32_16x16x32_bf16(false, a, false, b3, (short)0, acc3, false, false);
  }
  // C/D layout: lane holds column n0+lane&15; VGPR e holds row m0 + e + 8*(lane>=16)
  const int nl = lane & 15;
  const int mh = (lane >> 4) << 3;
  float* zr = Z + (size_t)(m0 + mh) * N4H + n0 + nl;
#pragma unroll
  for (int e = 0; e < 8; ++e) {
    zr[(size_t)e * N4H +  0] = acc0[e];
    zr[(size_t)e * N4H + 16] = acc1[e];
    zr[(size_t)e * N4H + 32] = acc2[e];
    zr[(size_t)e * N4H + 48] = acc3[e];
  }
}

// ---------------- generic bias GEMM: D[M][N] = A@W^T + bias ----------------
// Block = 128 thr = 4 waves along N; wave tile = 16x16. Grid = (M/16, N/64).
__global__ __launch_bounds__(128)
void gemm_bias_k(const __bf16* __restrict__ A, long lda,
                 const __bf16* __restrict__ W, long ldb,
                 const float* __restrict__ bias,
                 float* __restrict__ D, long ldd, int K) {
  const int lane = threadIdx.x & 31;
  const int wave = threadIdx.x >> 5;
  const int m0 = blockIdx.x * 16;
  const int n0 = blockIdx.y * 64 + wave * 16;
  const float bv = bias[n0 + (lane & 15)];
  f32x8 acc = {bv, bv, bv, bv, bv, bv, bv, bv};
  for (int k0 = 0; k0 < K; k0 += 32) {
    bf16x16 a = frag_a(A + (size_t)m0 * lda + k0, lda, lane);
    bf16x16 b = frag_b(W + (size_t)n0 * ldb + k0, ldb, lane);
    acc = __builtin_amdgcn_wmma_f32_16x16x32_bf16(false, a, false, b, (short)0, acc, false, false);
  }
  const int n = n0 + (lane & 15);
  const int mh = (lane >> 4) << 3;
#pragma unroll
  for (int e = 0; e < 8; ++e)
    D[(size_t)(m0 + mh + e) * ldd + n] = acc[e];
}

// ---------------- fused LSTM cell ------------------------------------------
__device__ __forceinline__ float sigm(float x) { return 1.0f / (1.0f + __expf(-x)); }

__global__ __launch_bounds__(256)
void lstm_cell_k(const float* __restrict__ Z, const float* __restrict__ bias,
                 float* __restrict__ c, __bf16* __restrict__ hstate,
                 float* __restrict__ dstF, long strideF,
                 __bf16* __restrict__ dstB, long strideB) {
  int idx = blockIdx.x * blockDim.x + threadIdx.x;
  if (idx >= BQ * HQ) return;
  int b = idx >> 9, j = idx & 511;
  const float* zr = Z + (size_t)b * N4H;
  float zi = zr[j]          + bias[j];
  float zf = zr[512 + j]    + bias[512 + j];
  float zg = zr[1024 + j]   + bias[1024 + j];
  float zo = zr[1536 + j]   + bias[1536 + j];
  float cn = sigm(zf) * c[idx] + sigm(zi) * tanhf(zg);
  float h  = sigm(zo) * tanhf(cn);
  c[idx] = cn;
  hstate[idx] = (__bf16)h;
  if (dstF) dstF[(size_t)b * strideF + j] = h;
  if (dstB) dstB[(size_t)b * strideB + j] = (__bf16)h;
}

// ---------------- conversion / fusion kernels ------------------------------
__global__ __launch_bounds__(256)
void cvt_pad_k(const float* __restrict__ src, int srck,
               __bf16* __restrict__ dst, int dstk, long nrows) {
  long idx = (long)blockIdx.x * blockDim.x + threadIdx.x;
  long total = nrows * (long)dstk;
  if (idx >= total) return;
  int  k = (int)(idx % dstk);
  long r = idx / dstk;
  float v = (k < srck) ? src[r * (long)srck + k] : 0.0f;
  dst[idx] = (__bf16)v;
}

__global__ __launch_bounds__(256)
void copy_rows_bf_k(__bf16* __restrict__ dst, long ldd, int off,
                    const __bf16* __restrict__ src) {
  int idx = blockIdx.x * blockDim.x + threadIdx.x;
  if (idx >= BQ * HQ) return;
  int b = idx >> 9, j = idx & 511;
  dst[(size_t)b * ldd + off + j] = src[idx];
}

// build glych[b][g][544] = concat(gly_0(16), gly_1 + (g==0)*pep_hx (512), pad 16)
__global__ __launch_bounds__(256)
void build_glych_k(__bf16* __restrict__ glych,
                   const float* __restrict__ gly0,
                   const float* __restrict__ gly1,
                   const float* __restrict__ pephx) {
  long idx = (long)blockIdx.x * blockDim.x + threadIdx.x;
  if (idx >= (long)BQ * GQ * 544) return;
  int  k  = (int)(idx % 544);
  long bg = idx / 544;
  int  g  = (int)(bg & (GQ - 1));
  long b  = bg >> 4;
  float v = 0.0f;
  if (k < 16) {
    v = gly0[bg * 16 + k];
  } else if (k < 528) {
    v = gly1[bg * 512 + (k - 16)];
    if (g == 0) v += pephx[b * 512 + (k - 16)];
  }
  glych[idx] = (__bf16)v;
}

// pep_gly[b][pos-1][:] += root_pep[b][:]  (in-place on bf16 pep_1 buffer)
__global__ __launch_bounds__(256)
void scatter_site_k(__bf16* __restrict__ pepgly,
                    const float* __restrict__ rootpep,
                    const int* __restrict__ gpos) {
  int idx = blockIdx.x * blockDim.x + threadIdx.x;
  if (idx >= BQ * 1024) return;
  int b = idx >> 10, j = idx & 1023;
  int pos = gpos[b] - 1;
  if (pos < 0) pos = 0;
  if (pos >= LQ) pos = LQ - 1;
  size_t off = ((size_t)b * LQ + pos) * 1024 + j;
  pepgly[off] = (__bf16)((float)pepgly[off] + rootpep[(size_t)b * 1024 + j]);
}

// ---------------------------------------------------------------------------
extern "C" void kernel_launch(void* const* d_in, const int* in_sizes, int n_in,
                              void* d_out, int out_size, void* d_ws, size_t ws_size,
                              hipStream_t stream) {
  (void)in_sizes; (void)n_in; (void)out_size; (void)ws_size;

  // ---- inputs (setup_inputs order) ----
  const float* pep0  = (const float*)d_in[0];   // [512,48,26]
  const float* gly0  = (const float*)d_in[1];   // [512,16,16]
  const int*   gpos  = (const int*)  d_in[2];   // [512]
  const float* p1Wih = (const float*)d_in[3];   // [2,2048,26]
  const float* p1Whh = (const float*)d_in[4];   // [2,2048,512]
  const float* p1b   = (const float*)d_in[5];   // [2,2048]
  const float* p2Wih = (const float*)d_in[6];   // [2,2048,1024]
  const float* p2Whh = (const float*)d_in[7];   // [2,2048,512]
  const float* p2b   = (const float*)d_in[8];   // [2,2048]
  const float* g1Wih = (const float*)d_in[9];   // [2048,16]
  const float* g1Whh = (const float*)d_in[10];  // [2048,512]
  const float* g1b   = (const float*)d_in[11];  // [2048]
  const float* g2Wih = (const float*)d_in[12];  // [2048,528]
  const float* g2Whh = (const float*)d_in[13];  // [2048,512]
  const float* g2b   = (const float*)d_in[14];  // [2048]
  const float* pgW   = (const float*)d_in[15];  // [512,1024]
  const float* pgb   = (const float*)d_in[16];  // [512]
  const float* gpW   = (const float*)d_in[17];  // [1024,512]
  const float* gpb   = (const float*)d_in[18];  // [1024]

  float* out      = (float*)d_out;                       // pep_2 [512,48,1024]
  float* out_gly2 = out + (size_t)BQ * LQ * 1024;        // gly_2 [8192,512]

  // ---- workspace carve-out ----
  char* wp = (char*)d_ws;
  auto carve = [&](size_t bytes) -> void* {
    void* r = (void*)wp;
    wp += (bytes + 255) & ~(size_t)255;
    return r;
  };
  __bf16* pepgly = (__bf16*)carve((size_t)BQ * LQ * 1024 * 2);  // pep_1/pep_gly (bf16)
  __bf16* glych  = (__bf16*)carve((size_t)BQ * GQ * 544 * 2);   // TreeLSTM2 input
  __bf16* pepx   = (__bf16*)carve((size_t)BQ * LQ * 32 * 2);    // pep_0 padded bf16
  __bf16* glyx   = (__bf16*)carve((size_t)BQ * GQ * 32 * 2);    // gly_0 padded bf16
  __bf16* w_p1ih = (__bf16*)carve((size_t)2 * N4H * 32 * 2);
  __bf16* w_p1hh = (__bf16*)carve((size_t)2 * N4H * HQ * 2);
  __bf16* w_p2ih = (__bf16*)carve((size_t)2 * N4H * 1024 * 2);
  __bf16* w_p2hh = (__bf16*)carve((size_t)2 * N4H * HQ * 2);
  __bf16* w_g1ih = (__bf16*)carve((size_t)N4H * 32 * 2);
  __bf16* w_g1hh = (__bf16*)carve((size_t)N4H * HQ * 2);
  __bf16* w_g2ih = (__bf16*)carve((size_t)N4H * 544 * 2);
  __bf16* w_g2hh = (__bf16*)carve((size_t)N4H * HQ * 2);
  __bf16* w_pg   = (__bf16*)carve((size_t)512 * 1024 * 2);
  __bf16* w_gp   = (__bf16*)carve((size_t)1024 * 512 * 2);
  __bf16* h_p1f  = (__bf16*)carve((size_t)BQ * HQ * 2);  // persists -> hTf
  __bf16* h_p1b  = (__bf16*)carve((size_t)BQ * HQ * 2);  // persists -> hTb
  __bf16* h_g1   = (__bf16*)carve((size_t)BQ * HQ * 2);  // persists -> gly root h
  __bf16* h_s    = (__bf16*)carve((size_t)BQ * HQ * 2);  // shared scratch state
  __bf16* hcat   = (__bf16*)carve((size_t)BQ * 1024 * 2);
  float*  cbuf   = (float*) carve((size_t)BQ * HQ * 4);
  float*  Zbuf   = (float*) carve((size_t)BQ * N4H * 4);
  float*  gly1f  = (float*) carve((size_t)BQ * GQ * HQ * 4);
  float*  pephx  = (float*) carve((size_t)BQ * HQ * 4);
  float*  rootpp = (float*) carve((size_t)BQ * 1024 * 4);

  // ---- helpers ----
  auto cvt = [&](const float* s, int sk, __bf16* d, int dk, long rows) {
    long tot = rows * (long)dk;
    cvt_pad_k<<<dim3((unsigned)((tot + 255) / 256)), 256, 0, stream>>>(s, sk, d, dk, rows);
  };
  auto gemm_step = [&](const __bf16* X, long ldx, const __bf16* Wx, int kx,
                       const __bf16* Hst, const __bf16* Wh) {
    lstm_step_gemm_k<<<dim3(16, 8), 256, 0, stream>>>(X, ldx, Wx, kx, Hst, Wh, Zbuf);
  };
  auto cell = [&](const float* bias, __bf16* hst,
                  float* dF, long sF, __bf16* dB, long sB) {
    lstm_cell_k<<<dim3(1024), 256, 0, stream>>>(Zbuf, bias, cbuf, hst, dF, sF, dB, sB);
  };
  auto zero_state = [&](__bf16* hst) {
    hipMemsetAsync(cbuf, 0, (size_t)BQ * HQ * 4, stream);
    hipMemsetAsync(hst, 0, (size_t)BQ * HQ * 2, stream);
  };

  // ---- 1. convert weights + inputs to bf16 (K padded to mult of 32) ----
  cvt(p1Wih,                26, w_p1ih,               32, N4H);
  cvt(p1Wih + (size_t)N4H * 26, 26, w_p1ih + (size_t)N4H * 32, 32, N4H);
  cvt(p1Whh, 512, w_p1hh, 512, 2L * N4H);
  cvt(p2Wih, 1024, w_p2ih, 1024, 2L * N4H);
  cvt(p2Whh, 512, w_p2hh, 512, 2L * N4H);
  cvt(g1Wih, 16, w_g1ih, 32, N4H);
  cvt(g1Whh, 512, w_g1hh, 512, N4H);
  cvt(g2Wih, 528, w_g2ih, 544, N4H);
  cvt(g2Whh, 512, w_g2hh, 512, N4H);
  cvt(pgW, 1024, w_pg, 1024, 512);
  cvt(gpW, 512, w_gp, 512, 1024);
  cvt(pep0, 26, pepx, 32, (long)BQ * LQ);
  cvt(gly0, 16, glyx, 32, (long)BQ * GQ);

  // ---- 2. BiLSTM1 (writes pep_1 in bf16 into pepgly) ----
  for (int dir = 0; dir < 2; ++dir) {
    __bf16* hst = dir ? h_p1b : h_p1f;
    zero_state(hst);
    const __bf16* Wih = w_p1ih + (size_t)dir * N4H * 32;
    const __bf16* Whh = w_p1hh + (size_t)dir * N4H * HQ;
    const float*  bia = p1b + (size_t)dir * N4H;
    for (int s = 0; s < LQ; ++s) {
      int t = dir ? (LQ - 1 - s) : s;
      gemm_step(pepx + (size_t)t * 32, (long)LQ * 32, Wih, 32, hst, Whh);
      cell(bia, hst, nullptr, 0, pepgly + (size_t)t * 1024 + dir * 512, (long)LQ * 1024);
    }
  }

  // ---- 3. TreeLSTM1: leaf(15) -> root(0); store h per node in gly1f ----
  zero_state(h_g1);
  for (int s = 0; s < GQ; ++s) {
    int g = GQ - 1 - s;
    gemm_step(glyx + (size_t)g * 32, (long)GQ * 32, w_g1ih, 32, h_g1, w_g1hh);
    cell(g1b, h_g1, gly1f + (size_t)g * HQ, (long)GQ * HQ, nullptr, 0);
  }
  // after loop: h_g1 holds node-0 (root) hidden in bf16

  // ---- 4. fusion: pep_hx = [hTf|hTb] @ pg_W^T + pg_b ----
  copy_rows_bf_k<<<dim3(1024), 256, 0, stream>>>(hcat, 1024, 0,   h_p1f);
  copy_rows_bf_k<<<dim3(1024), 256, 0, stream>>>(hcat, 1024, 512, h_p1b);
  gemm_bias_k<<<dim3(32, 8), 128, 0, stream>>>(hcat, 1024, w_pg, 1024, pgb, pephx, 512, 1024);
  //          root_pep = gly_1[:,0,:] @ gp_W^T + gp_b
  gemm_bias_k<<<dim3(32, 16), 128, 0, stream>>>(h_g1, 512, w_gp, 512, gpb, rootpp, 1024, 512);

  // ---- 5. build TreeLSTM2 input & scatter root_pep into peptide site ----
  build_glych_k<<<dim3((unsigned)(((long)BQ * GQ * 544 + 255) / 256)), 256, 0, stream>>>(
      glych, gly0, gly1f, pephx);
  scatter_site_k<<<dim3((BQ * 1024 + 255) / 256), 256, 0, stream>>>(pepgly, rootpp, gpos);

  // ---- 6. BiLSTM2 over pep_gly -> pep_2 (d_out part 1) ----
  for (int dir = 0; dir < 2; ++dir) {
    zero_state(h_s);
    const __bf16* Wih = w_p2ih + (size_t)dir * N4H * 1024;
    const __bf16* Whh = w_p2hh + (size_t)dir * N4H * HQ;
    const float*  bia = p2b + (size_t)dir * N4H;
    for (int s = 0; s < LQ; ++s) {
      int t = dir ? (LQ - 1 - s) : s;
      gemm_step(pepgly + (size_t)t * 1024, (long)LQ * 1024, Wih, 1024, h_s, Whh);
      cell(bia, h_s, out + (size_t)t * 1024 + dir * 512, (long)LQ * 1024, nullptr, 0);
    }
  }

  // ---- 7. TreeLSTM2: root(0) -> leaf(15) -> gly_2 (d_out part 2) ----
  zero_state(h_s);
  for (int g = 0; g < GQ; ++g) {
    gemm_step(glych + (size_t)g * 544, (long)GQ * 544, w_g2ih, 544, h_s, w_g2hh);
    cell(g2b, h_s, out_gly2 + (size_t)g * HQ, (long)GQ * HQ, nullptr, 0);
  }
}
